// DeepNet_22101901705918
// MI455X (gfx1250) — compile-verified, matrix operand order checked by reading, not run on hardware
//
#include <hip/hip_runtime.h>

#define HID 128

typedef __attribute__((ext_vector_type(16))) __bf16 v16bf;
typedef __attribute__((ext_vector_type(8)))  float  v8f;

__device__ __forceinline__ unsigned int pack_bf16x2(float a, float b) {
    // round-to-nearest-even fp32 -> bf16, packed (lo = first/even-k element)
    unsigned ua = __float_as_uint(a); ua = ua + 0x7FFFu + ((ua >> 16) & 1u);
    unsigned ub = __float_as_uint(b); ub = ub + 0x7FFFu + ((ub >> 16) & 1u);
    return (ua >> 16) | (ub & 0xFFFF0000u);
}

// ---------------------------------------------------------------------------
// Weight repack: W[k][n] fp32 row-major -> bf16 dwords in WMMA B-fragment
// per-lane layout: Bpack[ntile][kstep][lane][j], j=0..7 dwords per lane.
// Lane L: n = nt*16 + (L&15), hi = L>>4; dword j holds k-pair
//   k = ks*32 + (j&4 ? 16 : 0) + hi*8 + 2*(j&3).
// Two sources (w_root ksteps 0..kspersrc-1, w_nbr after) for the fused GEMM.
// ---------------------------------------------------------------------------
__global__ void repack_w(const float* __restrict__ W0, const float* __restrict__ W1,
                         unsigned int* __restrict__ Bpack, int kspersrc, int kstot) {
    int idx = blockIdx.x * blockDim.x + threadIdx.x;
    int per = kstot << 8;              // dwords per n-tile
    int total = 8 * per;               // 8 n-tiles cover HID=128
    if (idx >= total) return;
    int nt  = idx / per;
    int rem = idx - nt * per;
    int ks  = rem >> 8;
    int r2  = rem & 255;
    int L = r2 >> 3, j = r2 & 7;
    int hi = L >> 4;
    int n  = nt * 16 + (L & 15);
    int s   = ks / kspersrc;
    int ksl = ks - s * kspersrc;
    int k = ksl * 32 + ((j & 4) ? 16 : 0) + hi * 8 + 2 * (j & 3);
    const float* W = s ? W1 : W0;
    float v0 = W[k * HID + n];
    float v1 = W[(k + 1) * HID + n];
    Bpack[idx] = pack_bf16x2(v0, v1);
}

// ---------------------------------------------------------------------------
// Fused GEMM: out[m][n] = bias[n] + A0[m]@W0 (+ scale1[m]*A1[m]@W1)
// A sources fp32 [M][128]; Bpack prepacked bf16; fp32 accumulate via
// v_wmma_f32_16x16x32_bf16. 256 threads = 8 wave32; block tile = 32x128:
// each wave holds TWO 16x16 accumulators and reuses one B fragment per
// k-step for two WMMAs (halves weight traffic, doubles matrix-op density).
// A tiles converted fp32->bf16 into LDS in the WMMA A-fragment layout.
// Safe for out == A1 (tile rows fully read before written; no restrict).
// ---------------------------------------------------------------------------
__global__ __launch_bounds__(256)
void gemm_bf16_wmma(const float* __restrict__ A0, const float* A1,
                    const float* __restrict__ scale1,
                    const unsigned int* __restrict__ Bpack,
                    const float* __restrict__ bias, float* out,
                    int M, int kstot) {
    // apack[kstep_total][mtile(2)][lane(32)][j(8)] dwords: up to 8*512 = 16 KB
    __shared__ unsigned int apack[4096];
    const int tid = threadIdx.x;
    const int m0  = blockIdx.x * 32;

    // Prefetch the second source's rows while we pack the first
    // (emits global_prefetch_b8; speculative, dropped if invalid).
    if (A1 && tid < 32 && (m0 + tid) < M)
        __builtin_prefetch(A1 + (size_t)(m0 + tid) * HID, 0, 0);

    // ---- pack A tiles into LDS (4 ksteps = K128 per source, 2 m-tiles) ----
    const int nsrc = A1 ? 2 : 1;
    for (int idx = tid; idx < nsrc * 2048; idx += 256) {
        int rem = idx & 255;
        int L = rem >> 3, j = rem & 7;
        int blk = idx >> 8;            // (s, mt, ks) combined
        int ks  = blk & 3;
        int mt  = (blk >> 2) & 1;
        int s   = blk >> 3;
        const float* A  = s ? A1 : A0;
        const float* sc = s ? scale1 : nullptr;
        int kk  = ks * 32 + ((j & 4) ? 16 : 0) + ((L >> 4) << 3) + 2 * (j & 3);
        int row = m0 + mt * 16 + (L & 15);
        float v0 = 0.f, v1 = 0.f;
        if (row < M) {
            const float2 p = *(const float2*)(A + (size_t)row * HID + kk);
            float f = sc ? sc[row] : 1.0f;
            v0 = p.x * f; v1 = p.y * f;
        }
        apack[(s * 4 + ks) * 512 + mt * 256 + rem] = pack_bf16x2(v0, v1);
    }
    __syncthreads();

    const int wave = tid >> 5;
    const int lane = tid & 31;
    const int ncol = wave * 16 + (lane & 15);

    float bv = bias ? bias[ncol] : 0.f;
    v8f acc0, acc1;
#pragma unroll
    for (int r = 0; r < 8; ++r) { acc0[r] = bv; acc1[r] = bv; }

    const unsigned int* bp = Bpack + (size_t)(wave * kstot) * 256 + (lane << 3);
    for (int ks = 0; ks < kstot; ++ks) {
        union { uint4 q[2]; v16bf v; } fa0, fa1, fb;
        const uint4* bp4 = (const uint4*)(bp + ks * 256);
        fb.q[0] = bp4[0];
        fb.q[1] = bp4[1];
        const uint4* ap0 = (const uint4*)(apack + ks * 512 + (lane << 3));
        const uint4* ap1 = (const uint4*)(apack + ks * 512 + 256 + (lane << 3));
        fa0.q[0] = ap0[0]; fa0.q[1] = ap0[1];
        fa1.q[0] = ap1[0]; fa1.q[1] = ap1[1];
        acc0 = __builtin_amdgcn_wmma_f32_16x16x32_bf16(
            false, fa0.v, false, fb.v, (short)0, acc0, false, false);
        acc1 = __builtin_amdgcn_wmma_f32_16x16x32_bf16(
            false, fa1.v, false, fb.v, (short)0, acc1, false, false);
    }

    // C/D layout: VGPR r -> M = r (lanes 0-15) / r+8 (lanes 16-31); N = lane&15
    const int mbase = m0 + ((lane >> 4) << 3);
#pragma unroll
    for (int r = 0; r < 8; ++r) {
        int row0 = mbase + r;
        int row1 = row0 + 16;
        if (row0 < M) out[(size_t)row0 * HID + ncol] = acc0[r];
        if (row1 < M) out[(size_t)row1 * HID + ncol] = acc1[r];
    }
}

// ---------------------------------------------------------------------------
// Graph ops
// ---------------------------------------------------------------------------
__global__ void deg_count(const int* __restrict__ dst, float* __restrict__ deg, int E) {
    int t = blockIdx.x * blockDim.x + threadIdx.x;
    if (t < E) atomicAdd(&deg[dst[t]], 1.0f);
}

__global__ void inv_deg_k(float* __restrict__ deg, int N) {
    int t = blockIdx.x * blockDim.x + threadIdx.x;
    if (t < N) deg[t] = 1.0f / fmaxf(deg[t], 1.0f);
}

__global__ void scatter_edges(const float* __restrict__ node, const int* __restrict__ src,
                              const int* __restrict__ dst, float* __restrict__ agg, int E) {
    int t = blockIdx.x * blockDim.x + threadIdx.x;
    if (t >= E * 32) return;
    int e  = t >> 5;
    int c0 = (t & 31) * 4;
    int s = src[e], d = dst[e];
    const float4 v = *(const float4*)(node + (size_t)s * HID + c0);
    float* o = agg + (size_t)d * HID + c0;
    atomicAdd(o + 0, v.x); atomicAdd(o + 1, v.y);
    atomicAdd(o + 2, v.z); atomicAdd(o + 3, v.w);
}

// Fused: node = relu(LN(conv + node)) AND gconv[batch[row]] += conv[row]
// (graph pooling folded in so conv is streamed exactly once)
__global__ __launch_bounds__(HID)
void ln_relu_node_pool(const float* __restrict__ conv, float* __restrict__ node,
                       const int* __restrict__ batch, float* __restrict__ gconv,
                       const float* __restrict__ g, const float* __restrict__ b) {
    __shared__ float s1[HID], s2[HID];
    int row = blockIdx.x, c = threadIdx.x;
    size_t o = (size_t)row * HID + c;
    float vc = conv[o];
    atomicAdd(&gconv[(size_t)batch[row] * HID + c], vc);
    float v = vc + node[o];
    s1[c] = v; s2[c] = v * v;
    __syncthreads();
    for (int ofs = HID / 2; ofs > 0; ofs >>= 1) {
        if (c < ofs) { s1[c] += s1[c + ofs]; s2[c] += s2[c + ofs]; }
        __syncthreads();
    }
    float mean = s1[0] * (1.0f / HID);
    float var  = s2[0] * (1.0f / HID) - mean * mean;
    float y = (v - mean) * rsqrtf(var + 1e-5f) * g[c] + b[c];
    node[o] = fmaxf(y, 0.f);
}

// gact = relu(LN(gconv + (first ? 0 : gact)))
__global__ __launch_bounds__(HID)
void ln_relu_graph(const float* __restrict__ gconv, float* __restrict__ gact,
                   const float* __restrict__ g, const float* __restrict__ b, int first) {
    __shared__ float s1[HID], s2[HID];
    int row = blockIdx.x, c = threadIdx.x;
    size_t o = (size_t)row * HID + c;
    float v = gconv[o] + (first ? 0.f : gact[o]);
    s1[c] = v; s2[c] = v * v;
    __syncthreads();
    for (int ofs = HID / 2; ofs > 0; ofs >>= 1) {
        if (c < ofs) { s1[c] += s1[c + ofs]; s2[c] += s2[c + ofs]; }
        __syncthreads();
    }
    float mean = s1[0] * (1.0f / HID);
    float var  = s2[0] * (1.0f / HID) - mean * mean;
    float y = (v - mean) * rsqrtf(var + 1e-5f) * g[c] + b[c];
    gact[o] = fmaxf(y, 0.f);
}

// ---------------------------------------------------------------------------
extern "C" void kernel_launch(void* const* d_in, const int* in_sizes, int n_in,
                              void* d_out, int out_size, void* d_ws, size_t ws_size,
                              hipStream_t stream) {
    const float* x      = (const float*)d_in[0];
    const int*   ei     = (const int*)d_in[1];
    const int*   batch  = (const int*)d_in[2];
    const float* fc_w   = (const float*)d_in[3];
    const float* fc_b   = (const float*)d_in[4];
    const float* w_root = (const float*)d_in[5];
    const float* w_nbr  = (const float*)d_in[6];
    const float* bvec   = (const float*)d_in[7];
    const float* ln_g   = (const float*)d_in[8];
    const float* ln_b   = (const float*)d_in[9];

    const int N = in_sizes[0] / HID;
    const int E = in_sizes[1] / 2;
    const int G = (out_size - N * HID) / HID;
    const int* src = ei;
    const int* dst = ei + E;

    char* ws = (char*)d_ws;
    size_t off = 0;
    auto alloc = [&](size_t bytes) -> void* {
        void* p = ws + off;
        off += (bytes + 255) & ~(size_t)255;
        return p;
    };
    float*        node  = (float*)alloc((size_t)N * HID * sizeof(float));
    float*        agg   = (float*)alloc((size_t)N * HID * sizeof(float));  // reused as conv
    float*        deg   = (float*)alloc((size_t)N * sizeof(float));        // becomes inv_deg
    unsigned int* Bpack = (unsigned int*)alloc(8 * 8 * 256 * sizeof(unsigned int));
    float*        gconv = (float*)alloc((size_t)G * HID * sizeof(float));
    float*        gact  = (float*)alloc((size_t)G * HID * sizeof(float));

    // degree -> inv_deg (constant across blocks)
    hipMemsetAsync(deg, 0, (size_t)N * sizeof(float), stream);
    deg_count<<<(E + 255) / 256, 256, 0, stream>>>(dst, deg, E);
    inv_deg_k<<<(N + 255) / 256, 256, 0, stream>>>(deg, N);

    // input projection: node = x @ fc_w + fc_b   (K=128 -> 4 ksteps)
    repack_w<<<(8 * 4 * 256 + 255) / 256, 256, 0, stream>>>(fc_w, nullptr, Bpack, 4, 4);
    gemm_bf16_wmma<<<(N + 31) / 32, 256, 0, stream>>>(x, nullptr, nullptr, Bpack,
                                                      fc_b, node, N, 4);

    for (int i = 0; i < 3; ++i) {
        // agg = segment_sum(node[src], dst)  (inv_deg folded into GEMM A-pack)
        hipMemsetAsync(agg, 0, (size_t)N * HID * sizeof(float), stream);
        scatter_edges<<<(E * 32 + 255) / 256, 256, 0, stream>>>(node, src, dst, agg, E);

        // conv = node @ w_root[i] + (agg*inv_deg) @ w_nbr[i] + b[i]; in-place over agg
        repack_w<<<(8 * 8 * 256 + 255) / 256, 256, 0, stream>>>(
            w_root + (size_t)i * HID * HID, w_nbr + (size_t)i * HID * HID, Bpack, 4, 8);
        gemm_bf16_wmma<<<(N + 31) / 32, 256, 0, stream>>>(node, agg, deg, Bpack,
                                                          bvec + i * HID, agg, N, 8);

        // residual + LN + ReLU on nodes, with graph pooling fused in
        hipMemsetAsync(gconv, 0, (size_t)G * HID * sizeof(float), stream);
        ln_relu_node_pool<<<N, HID, 0, stream>>>(agg, node, batch, gconv,
                                                 ln_g + i * HID, ln_b + i * HID);
        ln_relu_graph<<<G, HID, 0, stream>>>(gconv, gact, ln_g + i * HID, ln_b + i * HID,
                                             i == 0 ? 1 : 0);
    }

    // outputs: [node (N*HID) | graph (G*HID)]
    hipMemcpyAsync(d_out, node, (size_t)N * HID * sizeof(float),
                   hipMemcpyDeviceToDevice, stream);
    hipMemcpyAsync((float*)d_out + (size_t)N * HID, gact, (size_t)G * HID * sizeof(float),
                   hipMemcpyDeviceToDevice, stream);
}
